// slot_dec_36240934044312
// MI455X (gfx1250) — matrix operand order repaired and checked by reading, not compiled, not run on hardware
//
#include <hip/hip_runtime.h>
#include <hip/hip_bf16.h>
#include <math.h>

typedef __attribute__((ext_vector_type(16))) _Float16 v16h;
typedef __attribute__((ext_vector_type(8)))  _Float16 v8h;
typedef __attribute__((ext_vector_type(8)))  float    v8f;

union V16u { v16h v; v8h h[2]; };

#define B_   64
#define L_   128
#define H_   512
#define H4_  2048
#define H5_  2560
#define LAB_ 128

__device__ __forceinline__ float4 ldf4(const float* p) {
  return *reinterpret_cast<const float4*>(p);
}

__device__ __forceinline__ v8h cvt8h(const float* p) {
  float4 a = ldf4(p);
  float4 b = ldf4(p + 4);
  v8h o;
  o[0] = (_Float16)a.x; o[1] = (_Float16)a.y;
  o[2] = (_Float16)a.z; o[3] = (_Float16)a.w;
  o[4] = (_Float16)b.x; o[5] = (_Float16)b.y;
  o[6] = (_Float16)b.z; o[7] = (_Float16)b.w;
  return o;
}

// ---- one-time fp32 -> f16 pack kernels ------------------------------------

__global__ __launch_bounds__(256)
void pack_x_f16(const float* __restrict__ x, const float* __restrict__ hi,
                _Float16* __restrict__ Xh) {
  const int m = blockIdx.x;
  const int t = m >> 6, b = m & 63;
  const int k = threadIdx.x * 8;
  const long base = (long)(b * L_ + t) * 1024;
  const float* src = (k < 1024) ? (x + base + k) : (hi + base + (k - 1024));
  *reinterpret_cast<v8h*>(Xh + (long)m * H4_ + k) = cvt8h(src);
}

__global__ void pack_w_f16(const float* __restrict__ W, _Float16* __restrict__ Wh,
                           int ld) {
  const int n = blockIdx.x;
  const int k = threadIdx.x * 8;
  if (k >= ld) return;
  *reinterpret_cast<v8h*>(Wh + (long)n * ld + k) = cvt8h(W + (long)n * ld + k);
}

// ---- pure-f16 WMMA GEMM, LDS-staged double-buffered B tile ----------------
// D[mBase+m, n] (+)= sum_k A[row(m), k] * Wh[n, KOFF+k]  (+bias0[n]+bias1[n])
// MT = 16-row m-subtiles per wave (1 or 2). MT=2 reuses each B fragment across
// two A fragments: 8 WMMAs per 8 ds_loads instead of 4.
template <int LDA, int LDW, int LDD, int K, int KOFF, int REMAP, int ACCUM, int MT>
__global__ __launch_bounds__(128)
void wmma_gemm_h(const _Float16* __restrict__ A,
                 const _Float16* __restrict__ Wh,
                 const float* __restrict__ bias0, const float* __restrict__ bias1,
                 float* __restrict__ D,
                 int M, int N, int mBase)
{
  const int tid  = threadIdx.x;
  const int lane = tid & 31;
  const int wave = tid >> 5;
  const int half = lane >> 4;   // ISA 7.12.2: lanes 0-15 vs 16-31 K-slice split
  const int lr   = lane & 15;

  const int m0 = blockIdx.x * (MT * 64) + wave * (MT * 16);
  const int n0 = blockIdx.y * 64;
  if (m0 >= M || n0 >= N) return;   // uniform (M multiple of MT*64, N of 64)

  const _Float16* arow[MT];
  #pragma unroll
  for (int mt = 0; mt < MT; ++mt) {
    int aRow = m0 + mt * 16 + lr;
    if (REMAP) { const int b = aRow >> 7, l = aRow & 127; aRow = l * B_ + b; }
    arow[mt] = A + (long)aRow * LDA;
  }

  // accumulators: MT x 4 tiles of 16x16 f32
  v8f acc[MT][4];
  #pragma unroll
  for (int mt = 0; mt < MT; ++mt) {
    #pragma unroll
    for (int nt = 0; nt < 4; ++nt) {
      const int ncol = n0 + nt * 16 + lr;
      if (ACCUM) {
        const float* dp = D + (long)(mBase + m0 + mt * 16 + half * 8) * LDD + ncol;
        #pragma unroll
        for (int r = 0; r < 8; ++r) acc[mt][nt][r] = dp[r * LDD];
      } else {
        float bi = 0.f;
        if (bias0) bi += bias0[ncol];
        if (bias1) bi += bias1[ncol];
        #pragma unroll
        for (int r = 0; r < 8; ++r) acc[mt][nt][r] = bi;
      }
    }
  }

  // B tile: 64 n-rows x 32 k f16 (4KB), double buffered in LDS.
  __shared__ _Float16 bbuf[2][64 * 32];
  const int bn = tid >> 1;
  const int bk = (tid & 1) * 16;
  const _Float16* wrow = Wh + (long)(n0 + bn) * LDW + KOFF + bk;

  v8h wr0, wr1;                 // global->LDS staging
  v8h a_c[MT][2], a_n[MT][2];   // A fragment double buffer

  wr0 = *reinterpret_cast<const v8h*>(wrow);
  wr1 = *reinterpret_cast<const v8h*>(wrow + 8);
  #pragma unroll
  for (int mt = 0; mt < MT; ++mt) {
    a_c[mt][0] = *reinterpret_cast<const v8h*>(arow[mt] + half * 8);
    a_c[mt][1] = *reinterpret_cast<const v8h*>(arow[mt] + 16 + half * 8);
  }
  *reinterpret_cast<v8h*>(&bbuf[0][bn * 32 + bk])     = wr0;
  *reinterpret_cast<v8h*>(&bbuf[0][bn * 32 + bk + 8]) = wr1;
  __syncthreads();

  constexpr int KT = K >> 5;
  for (int kt = 0; kt < KT; ++kt) {
    const int cur = kt & 1;
    if (kt + 1 < KT) {            // next tile's global loads go in flight now
      const int kn = (kt + 1) * 32;
      wr0 = *reinterpret_cast<const v8h*>(wrow + kn);
      wr1 = *reinterpret_cast<const v8h*>(wrow + kn + 8);
      #pragma unroll
      for (int mt = 0; mt < MT; ++mt) {
        a_n[mt][0] = *reinterpret_cast<const v8h*>(arow[mt] + kn + half * 8);
        a_n[mt][1] = *reinterpret_cast<const v8h*>(arow[mt] + kn + 16 + half * 8);
      }
    }
    if (kt + 2 < KT) {            // speculative prefetch two tiles ahead
      #pragma unroll
      for (int mt = 0; mt < MT; ++mt)
        __builtin_prefetch(arow[mt] + (kt + 2) * 32, 0, 1);
    }

    V16u au[MT];
    #pragma unroll
    for (int mt = 0; mt < MT; ++mt) { au[mt].h[0] = a_c[mt][0]; au[mt].h[1] = a_c[mt][1]; }

    // all four B fragments into distinct registers: ds_loads pipeline behind
    // one wait, then MT*4 WMMAs issue back-to-back.
    V16u bu[4];
    #pragma unroll
    for (int nt = 0; nt < 4; ++nt) {
      const _Float16* bp = &bbuf[cur][(nt * 16 + lr) * 32 + half * 8];
      bu[nt].h[0] = *reinterpret_cast<const v8h*>(bp);
      bu[nt].h[1] = *reinterpret_cast<const v8h*>(bp + 16);
    }
    #pragma unroll
    for (int nt = 0; nt < 4; ++nt) {
      #pragma unroll
      for (int mt = 0; mt < MT; ++mt) {
        acc[mt][nt] = __builtin_amdgcn_wmma_f32_16x16x32_f16(
            false, au[mt].v, false, bu[nt].v, (short)0, acc[mt][nt], false, false);
      }
    }

    if (kt + 1 < KT) {
      *reinterpret_cast<v8h*>(&bbuf[cur ^ 1][bn * 32 + bk])     = wr0;
      *reinterpret_cast<v8h*>(&bbuf[cur ^ 1][bn * 32 + bk + 8]) = wr1;
      #pragma unroll
      for (int mt = 0; mt < MT; ++mt) { a_c[mt][0] = a_n[mt][0]; a_c[mt][1] = a_n[mt][1]; }
    }
    __syncthreads();
  }

  #pragma unroll
  for (int mt = 0; mt < MT; ++mt) {
    #pragma unroll
    for (int nt = 0; nt < 4; ++nt) {
      const int ncol = n0 + nt * 16 + lr;
      float* dp = D + (long)(mBase + m0 + mt * 16 + half * 8) * LDD + ncol;
      #pragma unroll
      for (int r = 0; r < 8; ++r) dp[r * LDD] = acc[mt][nt][r];
    }
  }
}

// ---- irreducibly sequential LSTM-cell chain (fp32, one persistent WGP) ----
__global__ __launch_bounds__(1024)
void lstm_cell_seq(const float* __restrict__ PG_t,    // [64, 2048] pre-gates
                   const float* __restrict__ Whh,     // [2048, 512] fp32
                   _Float16* __restrict__ OUTh_t,     // [64, 512] f16 outputs
                   float* __restrict__ hc)            // persistent h|c
{
  __shared__ __align__(16) float h_s[H_];
  __shared__ __align__(16) float c_s[H_];
  __shared__ __align__(16) float g_s[H4_];
  const int tid = threadIdx.x;

  if (tid < H_) { h_s[tid] = hc[tid]; c_s[tid] = hc[H_ + tid]; }
  __syncthreads();

  for (int b = 0; b < B_; ++b) {
    const float* pg = PG_t + (long)b * H4_;
    #pragma unroll
    for (int hh = 0; hh < 2; ++hh) {
      const int n = tid + hh * 1024;
      const float* wr = Whh + (long)n * H_;
      float a0 = 0.f, a1 = 0.f;
      #pragma unroll 4
      for (int k = 0; k < H_; k += 8) {
        float4 w0 = ldf4(wr + k);
        float4 w1 = ldf4(wr + k + 4);
        float4 v0 = *reinterpret_cast<const float4*>(h_s + k);
        float4 v1 = *reinterpret_cast<const float4*>(h_s + k + 4);
        a0 = fmaf(w0.x, v0.x, a0); a0 = fmaf(w0.y, v0.y, a0);
        a0 = fmaf(w0.z, v0.z, a0); a0 = fmaf(w0.w, v0.w, a0);
        a1 = fmaf(w1.x, v1.x, a1); a1 = fmaf(w1.y, v1.y, a1);
        a1 = fmaf(w1.z, v1.z, a1); a1 = fmaf(w1.w, v1.w, a1);
      }
      g_s[n] = pg[n] + a0 + a1;
    }
    __syncthreads();

    if (tid < H_) {
      const int j = tid;
      const float ig = 1.f / (1.f + __expf(-g_s[j]));
      const float fg = 1.f / (1.f + __expf(-g_s[H_ + j]));
      const float gg = tanhf(g_s[2 * H_ + j]);
      const float og = 1.f / (1.f + __expf(-g_s[3 * H_ + j]));
      const float cn = fg * c_s[j] + ig * gg;
      const float hn = og * tanhf(cn);
      c_s[j] = cn;
      h_s[j] = hn;
      OUTh_t[(long)b * H_ + j] = (_Float16)hn;
    }
    __syncthreads();
  }

  if (tid < H_) { hc[tid] = h_s[tid]; hc[H_ + tid] = c_s[tid]; }
}

__global__ void zero_hc(float* hc) { hc[threadIdx.x] = 0.f; }

extern "C" void kernel_launch(void* const* d_in, const int* in_sizes, int n_in,
                              void* d_out, int out_size, void* d_ws, size_t ws_size,
                              hipStream_t stream) {
  (void)in_sizes; (void)n_in; (void)out_size; (void)ws_size;
  const float* x    = (const float*)d_in[0];
  const float* hi   = (const float*)d_in[1];
  const float* W_ih = (const float*)d_in[2];
  const float* W_hh = (const float*)d_in[3];
  const float* b_ih = (const float*)d_in[4];
  const float* b_hh = (const float*)d_in[5];
  const float* W_fc = (const float*)d_in[6];
  const float* b_fc = (const float*)d_in[7];
  float* out = (float*)d_out;

  float* ws = (float*)d_ws;
  float* HC = ws;                                   // h[512] | c[512]
  float* PG = ws + 1024;                            // [8192, 2048] fp32 (64 MB)
  _Float16* Xh    = (_Float16*)(PG + (size_t)L_ * B_ * H4_);  // [8192,2048] f16
  _Float16* Wih_h = Xh + (size_t)L_ * B_ * H4_;     // [2048, 2560] f16
  _Float16* Wfc_h = Wih_h + (size_t)H4_ * H5_;      // [128, 512]   f16
  _Float16* OUTh  = Wfc_h + (size_t)LAB_ * H_;      // [8192, 512]  f16

  zero_hc<<<1, 1024, 0, stream>>>(HC);
  pack_x_f16<<<L_ * B_, 256, 0, stream>>>(x, hi, Xh);
  pack_w_f16<<<H4_, 320, 0, stream>>>(W_ih, Wih_h, H5_);
  pack_w_f16<<<LAB_, 64, 0, stream>>>(W_fc, Wfc_h, H_);

  // Phase 1: PG = Xh @ W_ih[:, :2048]^T + b_ih + b_hh   (128x64 block tile)
  wmma_gemm_h<H4_, H5_, H4_, H4_, 0, 0, 0, 2>
      <<<dim3(L_ * B_ / 128, H4_ / 64), 128, 0, stream>>>(
          Xh, Wih_h, b_ih, b_hh, PG, L_ * B_, H4_, 0);

  // Phase 2+3: per-timestep recurrent-input GEMM + sequential cell chain
  for (int t = 0; t < L_; ++t) {
    if (t > 0) {
      wmma_gemm_h<H_, H5_, H4_, H_, H4_, 0, 1, 1>
          <<<dim3(1, H4_ / 64), 128, 0, stream>>>(
              OUTh + (size_t)(t - 1) * B_ * H_, Wih_h,
              nullptr, nullptr, PG, B_, H4_, t * B_);
    }
    lstm_cell_seq<<<1, 1024, 0, stream>>>(
        PG + (size_t)t * B_ * H4_, W_hh, OUTh + (size_t)t * B_ * H_, HC);
  }

  // Phase 4: out[b*128+l, :] = OUTh[l*64+b, :] @ W_fc^T + b_fc
  wmma_gemm_h<H_, H_, LAB_, H_, 0, 1, 0, 2>
      <<<dim3(L_ * B_ / 128, LAB_ / 64), 128, 0, stream>>>(
          OUTh, Wfc_h, b_fc, nullptr, out, L_ * B_, LAB_, 0);
}
